// GNNSimpleLP_16123307229265
// MI455X (gfx1250) — matrix operand order repaired and checked by reading, not compiled, not run on hardware
//
#include <hip/hip_runtime.h>

typedef __attribute__((ext_vector_type(16))) _Float16 v16h;
typedef __attribute__((ext_vector_type(8)))  _Float16 v8h;
typedef __attribute__((ext_vector_type(4)))  _Float16 v4h;
typedef __attribute__((ext_vector_type(8)))  float    v8f;
typedef __attribute__((ext_vector_type(4)))  float    v4f;

#define D 64

// ---------- degree / norm ----------
__global__ void k_set1(float* __restrict__ p, int n) {
  int i = blockIdx.x * blockDim.x + threadIdx.x;
  if (i < n) p[i] = 1.0f;                       // self-loop contributes 1 to degree
}

__global__ void k_deg_count(const int* __restrict__ dst, float* __restrict__ deg, int E) {
  int i = blockIdx.x * blockDim.x + threadIdx.x;
  if (i < E) atomicAdd(&deg[dst[i]], 1.0f);
}

__global__ void k_rsqrt(float* __restrict__ p, int n) {
  int i = blockIdx.x * blockDim.x + threadIdx.x;
  if (i < n) p[i] = rsqrtf(p[i]);
}

// ---------- precision conversion (4-wide: b128 load -> b64 store) ----------
__global__ void k_cast_f16(const float* __restrict__ in, _Float16* __restrict__ out,
                           int ngroups) {
  int i = blockIdx.x * blockDim.x + threadIdx.x;
  if (i >= ngroups) return;
  const v4f v = *(const v4f*)(in + (size_t)i * 4);
  v4h o;
#pragma unroll
  for (int j = 0; j < 4; ++j) o[j] = (_Float16)v[j];
  *(v4h*)(out + (size_t)i * 4) = o;
}

// Weights -> f16, transposed (wt[n][k] = W[k][n]) so each B fragment is one
// contiguous 32B load.
__global__ void k_prep_wt(const float* __restrict__ W1, const float* __restrict__ W2,
                          const float* __restrict__ Wp, _Float16* __restrict__ wt1,
                          _Float16* __restrict__ wt2, _Float16* __restrict__ wtp) {
  const float* W = (blockIdx.x == 0) ? W1 : (blockIdx.x == 1) ? W2 : Wp;
  _Float16*    o = (blockIdx.x == 0) ? wt1 : (blockIdx.x == 1) ? wt2 : wtp;
  for (int i = threadIdx.x; i < D * D; i += blockDim.x) {
    int n = i / D, k = i % D;
    o[n * D + k] = (_Float16)W[k * D + n];
  }
}

// ---------- fused GEMM ----------
// One wave = one 16-row stripe x all 64 columns: 4 n-tiles, K=64 as two
// k-steps -> 8 v_wmma_f32_16x16x32_f16 on 4 independent accumulators.
// PROJ==0: store hw = A@B  and  agg = inv[row]^2 * (A@B)   (self-loop init fused)
// PROJ==1: store out = A@B + bias                           (projection)
#define WMMA16(a, b, c) \
  __builtin_amdgcn_wmma_f32_16x16x32_f16(false, (a), false, (b), (short)0, (c), false, false)

template <int PROJ>
__global__ void k_gemm64_fused(const _Float16* __restrict__ A, const _Float16* __restrict__ Bt,
                               const float* __restrict__ bias, const float* __restrict__ inv,
                               float* __restrict__ hw, float* __restrict__ agg, int M) {
  const int lane = threadIdx.x & 31;
  const int wave = threadIdx.x >> 5;
  const int mt = blockIdx.x * (blockDim.x >> 5) + wave;
  if (mt * 16 >= M) return;                     // wave-uniform (M % 16 == 0)
  const int q   = lane >> 4;                    // lane half
  const int loc = lane & 15;

  // A fragments (16x32 each): lane = row mt*16+loc, K = kk + q*16 + 0..15
  const _Float16* arow = A + (size_t)(mt * 16 + loc) * D;
  v16h a0, a1;
  {
    v8h t0 = *(const v8h*)(arow + q * 16);
    v8h t1 = *(const v8h*)(arow + 8 + q * 16);
    v8h t2 = *(const v8h*)(arow + 32 + q * 16);
    v8h t3 = *(const v8h*)(arow + 40 + q * 16);
#pragma unroll
    for (int i = 0; i < 8; ++i) {
      a0[i] = t0[i]; a0[i + 8] = t1[i];
      a1[i] = t2[i]; a1[i + 8] = t3[i];
    }
  }
  // B fragments (32x16): lane = column nt*16+loc, K = kk + q*16 + 0..15
  const _Float16* bbase = Bt + (size_t)loc * D + q * 16;
  v16h b00 = *(const v16h*)(bbase + 0 * 16 * D);
  v16h b10 = *(const v16h*)(bbase + 1 * 16 * D);
  v16h b20 = *(const v16h*)(bbase + 2 * 16 * D);
  v16h b30 = *(const v16h*)(bbase + 3 * 16 * D);
  v16h b01 = *(const v16h*)(bbase + 0 * 16 * D + 32);
  v16h b11 = *(const v16h*)(bbase + 1 * 16 * D + 32);
  v16h b21 = *(const v16h*)(bbase + 2 * 16 * D + 32);
  v16h b31 = *(const v16h*)(bbase + 3 * 16 * D + 32);

  v8f c0 = {}, c1 = {}, c2 = {}, c3 = {};
  c0 = WMMA16(a0, b00, c0);
  c1 = WMMA16(a0, b10, c1);
  c2 = WMMA16(a0, b20, c2);
  c3 = WMMA16(a0, b30, c3);
  c0 = WMMA16(a1, b01, c0);
  c1 = WMMA16(a1, b11, c1);
  c2 = WMMA16(a1, b21, c2);
  c3 = WMMA16(a1, b31, c3);

  // D layout: VGPR r -> row q*8 + r, column loc (+16 per n-tile).
  // All stores from one base + immediate offsets (straight-line clause).
  const int row0 = mt * 16 + q * 8;
  float* o = hw + (size_t)row0 * D + loc;
  if (PROJ) {
    const float bv0 = bias[loc];
    const float bv1 = bias[loc + 16];
    const float bv2 = bias[loc + 32];
    const float bv3 = bias[loc + 48];
#pragma unroll
    for (int r = 0; r < 8; ++r) {
      o[r * D]      = c0[r] + bv0;
      o[r * D + 16] = c1[r] + bv1;
      o[r * D + 32] = c2[r] + bv2;
      o[r * D + 48] = c3[r] + bv3;
    }
  } else {
    float* g = agg + (size_t)row0 * D + loc;
    const float* ip = inv + row0;
#pragma unroll
    for (int r = 0; r < 8; ++r) {
      float w = ip[r]; w = w * w;               // inv_sqrt_deg^2 (self-loop norm)
      o[r * D]      = c0[r];  g[r * D]      = w * c0[r];
      o[r * D + 16] = c1[r];  g[r * D + 16] = w * c1[r];
      o[r * D + 32] = c2[r];  g[r * D + 32] = w * c2[r];
      o[r * D + 48] = c3[r];  g[r * D + 48] = w * c3[r];
    }
  }
}

// ---------- edge scatter ----------
// One wave per edge: 32 lanes x float2 = 64 features, coalesced 256B gather +
// 64 non-returning global_atomic_add_f32 resolving in the 192MB L2.
__global__ void k_scatter(const float* __restrict__ hw, const int* __restrict__ src,
                          const int* __restrict__ dst, const float* __restrict__ inv,
                          float* __restrict__ agg, int E) {
  const int lane = threadIdx.x & 31;
  const int e = (blockIdx.x * blockDim.x + threadIdx.x) >> 5;
  if (e >= E) return;
  const int s = src[e], d = dst[e];
  const float norm = inv[s] * inv[d];
  const float2 v = *((const float2*)(hw + (size_t)s * D) + lane);
  float* o = agg + (size_t)d * D + lane * 2;
  atomicAdd(o,     norm * v.x);
  atomicAdd(o + 1, norm * v.y);
}

// h = relu(agg + b), cast to f16; 4-wide (b128 load -> b64 store)
__global__ void k_bias_relu_cast(const float* __restrict__ agg, const float* __restrict__ b,
                                 _Float16* __restrict__ out, int ngroups) {
  int i = blockIdx.x * blockDim.x + threadIdx.x;
  if (i >= ngroups) return;
  const size_t base = (size_t)i * 4;
  const v4f v  = *(const v4f*)(agg + base);
  const v4f bv = *(const v4f*)(b + (base & (D - 1)));   // D%4==0 -> contiguous
  v4h o;
#pragma unroll
  for (int j = 0; j < 4; ++j) o[j] = (_Float16)fmaxf(v[j] + bv[j], 0.0f);
  *(v4h*)(out + base) = o;
}

extern "C" void kernel_launch(void* const* d_in, const int* in_sizes, int n_in,
                              void* d_out, int out_size, void* d_ws, size_t ws_size,
                              hipStream_t stream) {
  const float* x  = (const float*)d_in[0];
  const int*   ei = (const int*)d_in[1];
  const float* W1 = (const float*)d_in[2];
  const float* b1 = (const float*)d_in[3];
  const float* W2 = (const float*)d_in[4];
  const float* b2 = (const float*)d_in[5];
  const float* Wp = (const float*)d_in[6];
  const float* bp = (const float*)d_in[7];

  const int N = in_sizes[0] / D;
  const int E = in_sizes[1] / 2;
  const int* src = ei;
  const int* dst = ei + E;
  const int total  = N * D;
  const int groups = total / 4;                       // D%4==0

  // workspace carve-out (~64.5 MB; fits MI455X 192MB L2)
  char* w = (char*)d_ws;
  size_t off = 0;
  auto carve = [&](size_t bytes) -> char* {
    char* p = w + off;
    off = (off + bytes + 255) & ~(size_t)255;
    return p;
  };
  float*    inv = (float*)carve((size_t)N * 4);
  float*    agg = (float*)carve((size_t)total * 4);
  float*    hw  = (float*)carve((size_t)total * 4);
  _Float16* hh  = (_Float16*)carve((size_t)total * 2);
  _Float16* wt1 = (_Float16*)carve(D * D * 2);
  _Float16* wt2 = (_Float16*)carve(D * D * 2);
  _Float16* wtp = (_Float16*)carve(D * D * 2);
  (void)ws_size; (void)n_in; (void)out_size;

  const int T = 256;                                  // 8 wave32 per block
  const int gN = (N + T - 1) / T;
  const int gE = (E + T - 1) / T;
  const int gG = (groups + T - 1) / T;
  const int gemm_blocks = ((N + 15) / 16 + 7) / 8;    // one wave per 16-row stripe
  const int scat_blocks = (int)(((long long)E * 32 + T - 1) / T);

  k_prep_wt<<<3, T, 0, stream>>>(W1, W2, Wp, wt1, wt2, wtp);
  k_set1<<<gN, T, 0, stream>>>(inv, N);
  k_deg_count<<<gE, T, 0, stream>>>(dst, inv, E);
  k_rsqrt<<<gN, T, 0, stream>>>(inv, N);
  k_cast_f16<<<gG, T, 0, stream>>>(x, hh, groups);

  // ---- layer 1 ----
  k_gemm64_fused<0><<<gemm_blocks, T, 0, stream>>>(hh, wt1, nullptr, inv, hw, agg, N);
  k_scatter<<<scat_blocks, T, 0, stream>>>(hw, src, dst, inv, agg, E);
  k_bias_relu_cast<<<gG, T, 0, stream>>>(agg, b1, hh, groups);

  // ---- layer 2 ----
  k_gemm64_fused<0><<<gemm_blocks, T, 0, stream>>>(hh, wt2, nullptr, inv, hw, agg, N);
  k_scatter<<<scat_blocks, T, 0, stream>>>(hw, src, dst, inv, agg, E);
  k_bias_relu_cast<<<gG, T, 0, stream>>>(agg, b2, hh, groups);

  // ---- projection (bias fused) ----
  k_gemm64_fused<1><<<gemm_blocks, T, 0, stream>>>(hh, wtp, bp, nullptr, (float*)d_out, nullptr, N);
}
